// MultiHeadAttention_84602265797230
// MI455X (gfx1250) — matrix operand order repaired and checked by reading, not compiled, not run on hardware
//
#include <hip/hip_runtime.h>
#include <stdint.h>

// ---------------------------------------------------------------------------
// MI455X (gfx1250, wave32) multi-head attention forward.
// GEMMs via v_wmma_f32_16x16x32_bf16; bf16 tiles staged to LDS with the
// Tensor Data Mover (tensor_load_to_lds), double-buffered in the attention
// kernel (issue stage s+1, s_wait_tensorcnt 2 -> stage s complete).
// ---------------------------------------------------------------------------

typedef __attribute__((ext_vector_type(16))) __bf16 v16bf;
typedef __attribute__((ext_vector_type(8)))  float  v8f;
typedef __attribute__((ext_vector_type(4)))  unsigned int u32x4;
typedef __attribute__((ext_vector_type(8)))  int    i32x8;
typedef __attribute__((ext_vector_type(4)))  int    i32x4;

#define WMMA_BF16(a, b, c)                                                     \
  __builtin_amdgcn_wmma_f32_16x16x32_bf16(false, (a), false, (b), (short)0,    \
                                          (c), false, false)

#ifndef __has_builtin
#define __has_builtin(x) 0
#endif
#if __has_builtin(__builtin_amdgcn_tensor_load_to_lds) &&                      \
    __has_builtin(__builtin_amdgcn_s_wait_tensorcnt)
#define USE_TDM 1
#else
#define USE_TDM 0
#endif

constexpr int BB = 2;
constexpr int SS = 2048;
constexpr int DD = 1024;
constexpr int HH = 16;
constexpr int HD = 64;
constexpr int MTOT = BB * SS;   // 4096

// ---------------- WMMA fragment loaders (contiguous k = 16*half + e order) ----------------

// A matrix 16x32 bf16: lane = {m = lane&15, half = lane>>4};
// element e -> k = half*16 + e
__device__ __forceinline__ v16bf load_a_frag(const __bf16* base, int ld) {
  const int lane = threadIdx.x & 31;
  const int m  = lane & 15;
  const int hl = (lane >> 4) & 1;
  v16bf a;
#pragma unroll
  for (int e = 0; e < 16; ++e) {
    const int k = hl * 16 + e;
    a[e] = base[m * ld + k];
  }
  return a;
}

// B matrix 32x16 bf16, row-major (k-major) with stride ld:
// lane = {n = lane&15, half = lane>>4}; element e -> k = half*16 + e
__device__ __forceinline__ v16bf load_b_frag(const __bf16* base, int ld) {
  const int lane = threadIdx.x & 31;
  const int n  = lane & 15;
  const int hl = (lane >> 4) & 1;
  v16bf b;
#pragma unroll
  for (int e = 0; e < 16; ++e) {
    const int k = hl * 16 + e;
    b[e] = base[k * ld + n];
  }
  return b;
}

// B fragment where logical B(k,n) lives at base[n*ld + k] (reading a
// row-major K matrix as K^T for Q*K^T). Per-lane contiguous (2x b128).
__device__ __forceinline__ v16bf load_bT_frag(const __bf16* base, int ld) {
  const int lane = threadIdx.x & 31;
  const int n  = lane & 15;
  const int hl = (lane >> 4) & 1;
  v16bf b;
#pragma unroll
  for (int e = 0; e < 16; ++e) {
    const int k = hl * 16 + e;
    b[e] = base[n * ld + k];
  }
  return b;
}

// ---------------- Tensor Data Mover: 2D bf16 tile -> LDS --------------------
// rows x cols bf16 tile, global row stride rstride (elements); LDS rows get
// (pad_amt+1) DWORDs of padding after every 2<<pad_int DWORDs (= one row).

#if USE_TDM
__device__ __forceinline__ void tdm_load_2d(const void* gptr, const void* lptr,
                                            int cols, int rows, int rstride,
                                            int pad_int, int pad_amt) {
  const unsigned long long ga = (unsigned long long)(uintptr_t)gptr;
  const unsigned lds = (unsigned)(uintptr_t)lptr;  // low 32 bits = LDS offset
  u32x4 g0 = {
      1u,                                        // count=1, user, no gather
      lds,                                       // lds_addr        [63:32]
      (unsigned)(ga & 0xFFFFFFFFu),              // global_addr lo  [95:64]
      (unsigned)((ga >> 32) & 0x01FFFFFFu) |     // global_addr hi  [120:96]
          (2u << 30)                             // type=2 (image)  [127:126]
  };
  i32x8 g1 = {
      (1 << 16) |                                // data_size = 2 bytes
          (1 << 20) |                            // pad_enable
          (pad_int << 22) | (pad_amt << 25),     // LDS row padding
      (cols & 0xFFFF) << 16,                     // tensor_dim0
      (rows & 0xFFFF) << 16,                     // tensor_dim1
      (cols & 0xFFFF) << 16,                     // tile_dim0
      rows & 0xFFFF,                             // tile_dim1
      rstride,                                   // tensor_dim0_stride
      0, 0};
  i32x4 gz = {0, 0, 0, 0};
#if defined(__clang_major__) && (__clang_major__ >= 23)
  i32x8 gz8 = {0, 0, 0, 0, 0, 0, 0, 0};
  __builtin_amdgcn_tensor_load_to_lds(g0, g1, gz, gz, gz8, 0);
#else
  __builtin_amdgcn_tensor_load_to_lds(g0, g1, gz, gz, 0);
#endif
}
#endif

// ---------------- GEMM + bias:  C[M,N] = A[M,K] * W[K,N] + bias ------------

constexpr int TM = 128, TN = 128, TK = 32;
constexpr int LDA = TK + 2;   // 34 = 16 DWORDs data + 1 DWORD pad (TDM layout)
constexpr int LDB = TN + 2;

template <typename AT, typename OT>
__global__ __launch_bounds__(256) void gemm_bias_kernel(
    const AT* __restrict__ A, const float* __restrict__ W,
    const float* __restrict__ bias, OT* __restrict__ C, int M, int N, int K) {
  __shared__ __bf16 sA[TM * LDA];
  __shared__ __bf16 sB[TK * LDB];

  const int t    = threadIdx.x;
  const int wid  = t >> 5;
  const int lane = t & 31;
  const int m0 = blockIdx.x * TM;
  const int n0 = blockIdx.y * TN;
  const int waveM = (wid >> 2) * 64;
  const int waveN = (wid & 3) * 32;

  v8f acc[4][2];
#pragma unroll
  for (int i = 0; i < 4; ++i)
#pragma unroll
    for (int j = 0; j < 2; ++j) acc[i][j] = {};

  const int arow = t >> 1, acol = (t & 1) * 16;  // A stage: 128 x 32
  const int brow = t >> 3, bcol = (t & 7) * 16;  // W stage: 32 x 128

#if USE_TDM
  const bool tdmA   = (sizeof(AT) == 2);         // A already bf16 -> DMA it
  const bool leader = (__builtin_amdgcn_readfirstlane(wid) == 0);
#endif

  for (int kk = 0; kk < K; kk += TK) {
#if USE_TDM
    if (tdmA) {
      if (leader)  // 128x32 tile, LDS stride 34 bf16 (16 DWORDs + 1 pad)
        tdm_load_2d(A + (size_t)m0 * K + kk, sA, TK, TM, K, 3, 0);
    } else
#endif
    {
      const AT* src = A + (size_t)(m0 + arow) * K + kk + acol;
#pragma unroll
      for (int e = 0; e < 16; ++e)
        sA[arow * LDA + acol + e] = (__bf16)(float)src[e];
    }
    {
      const float* src = W + (size_t)(kk + brow) * N + n0 + bcol;
#pragma unroll
      for (int e = 0; e < 16; ++e)
        sB[brow * LDB + bcol + e] = (__bf16)src[e];
    }
    if (kk + TK < K)
      __builtin_prefetch(W + (size_t)(kk + TK + brow) * N + n0 + bcol, 0, 1);
#if USE_TDM
    if (tdmA && leader) __builtin_amdgcn_s_wait_tensorcnt((short)0);
#endif
    __syncthreads();

    v16bf bf[2];
#pragma unroll
    for (int j = 0; j < 2; ++j)
      bf[j] = load_b_frag(&sB[waveN + j * 16], LDB);
#pragma unroll
    for (int i = 0; i < 4; ++i) {
      v16bf af = load_a_frag(&sA[(waveM + i * 16) * LDA], LDA);
#pragma unroll
      for (int j = 0; j < 2; ++j) acc[i][j] = WMMA_BF16(af, bf[j], acc[i][j]);
    }
    __syncthreads();
  }

  const int nloc  = lane & 15;
  const int mhalf = (lane >> 4) * 8;
#pragma unroll
  for (int j = 0; j < 2; ++j) {
    const int n  = n0 + waveN + j * 16 + nloc;
    const float bj = bias[n];
#pragma unroll
    for (int i = 0; i < 4; ++i) {
      const int mb = m0 + waveM + i * 16 + mhalf;
#pragma unroll
      for (int r = 0; r < 8; ++r)
        C[(size_t)(mb + r) * N + n] = (OT)(acc[i][j][r] + bj);
    }
  }
}

// ---------------- Flash attention: O = softmax(Q K^T / 8) V ----------------
// One workgroup per (128 q-rows, b, h); 8 waves, 16 q-rows per wave.
// K/V tiles double-buffered in LDS: leader wave issues the TDM for stage s+1,
// then s_wait_tensorcnt 2 guarantees stage s has landed (in-order per wave).

constexpr int QB  = 128;
constexpr int LDH = HD + 8;   // 72 (TDM pad: 32 DWORDs data + 4 DWORDs pad)
constexpr int NS  = SS / 32;  // 64 key stages

__global__ __launch_bounds__(256) void attention_kernel(
    const __bf16* __restrict__ Qp, const __bf16* __restrict__ Kp,
    const __bf16* __restrict__ Vp, __bf16* __restrict__ O) {
  __shared__ __bf16 sK[2][32 * LDH];
  __shared__ __bf16 sV[2][32 * LDH];
  __shared__ __bf16 sP[8 * 16 * 34];  // per-wave P re-layout scratch

  const int t    = threadIdx.x;
  const int wid  = t >> 5;
  const int lane = t & 31;
  const int nloc = lane & 15;
  const int hl   = (lane >> 4) & 1;

  const int b = blockIdx.y >> 4;
  const int h = blockIdx.y & 15;
  const int hoff    = h * HD;
  const int rowbase = b * SS;
  const int qrow0   = blockIdx.x * QB + wid * 16;

  const __bf16* Kh = Kp + (size_t)rowbase * DD + hoff;
  const __bf16* Vh = Vp + (size_t)rowbase * DD + hoff;

#if USE_TDM
  const bool leader = (__builtin_amdgcn_readfirstlane(wid) == 0);
  if (leader) {  // prologue: stage 0 into buffer 0
    tdm_load_2d(Kh, sK[0], HD, 32, DD, 4, 3);
    tdm_load_2d(Vh, sV[0], HD, 32, DD, 4, 3);
  }
#else
  const int srow = t >> 3, scol = (t & 7) * 8;  // 32 x 64 staging fallback
#endif

  // Preload Q fragments (A layout, straight from global; overlaps the DMA)
  v16bf qa[2];
  {
    const __bf16* qbase =
        Qp + (size_t)(rowbase + qrow0 + (lane & 15)) * DD + hoff;
#pragma unroll
    for (int c = 0; c < 2; ++c)
#pragma unroll
      for (int e = 0; e < 16; ++e) {
        const int k = hl * 16 + e;
        qa[c][e] = qbase[c * 32 + k];
      }
  }

  v8f o[4];
#pragma unroll
  for (int j = 0; j < 4; ++j) o[j] = {};
  float rowmax[8], rowsum[8];
#pragma unroll
  for (int r = 0; r < 8; ++r) { rowmax[r] = -3.0e38f; rowsum[r] = 0.0f; }

  const float kscale = 0.125f;  // 1/sqrt(64)

  for (int s = 0; s < NS; ++s) {
    const int cur = s & 1;
#if USE_TDM
    if (leader) {
      if (s + 1 < NS) {  // prefetch stage s+1 into the other buffer
        tdm_load_2d(Kh + (size_t)(s + 1) * 32 * DD, sK[cur ^ 1], HD, 32, DD, 4, 3);
        tdm_load_2d(Vh + (size_t)(s + 1) * 32 * DD, sV[cur ^ 1], HD, 32, DD, 4, 3);
        __builtin_amdgcn_s_wait_tensorcnt((short)2);  // stage s landed
      } else {
        __builtin_amdgcn_s_wait_tensorcnt((short)0);
      }
    }
#else
    {
      const size_t g = (size_t)(s * 32 + srow) * DD + scol;
#pragma unroll
      for (int e = 0; e < 8; ++e) sK[cur][srow * LDH + scol + e] = Kh[g + e];
#pragma unroll
      for (int e = 0; e < 8; ++e) sV[cur][srow * LDH + scol + e] = Vh[g + e];
    }
#endif
    __syncthreads();

    // ---- scores: two 16x16 tiles (keys 32s..+15, +16..+31), K-dim = 64 ----
    v8f s0 = {}, s1 = {};
    s0 = WMMA_BF16(qa[0], load_bT_frag(&sK[cur][0 * 16 * LDH + 0], LDH), s0);
    s0 = WMMA_BF16(qa[1], load_bT_frag(&sK[cur][0 * 16 * LDH + 32], LDH), s0);
    s1 = WMMA_BF16(qa[0], load_bT_frag(&sK[cur][1 * 16 * LDH + 0], LDH), s1);
    s1 = WMMA_BF16(qa[1], load_bT_frag(&sK[cur][1 * 16 * LDH + 32], LDH), s1);

    // ---- online softmax (rows live on 16-lane halves; xor<=8 stays inside)
#pragma unroll
    for (int r = 0; r < 8; ++r) {
      float a0 = s0[r] * kscale, a1 = s1[r] * kscale;
      float mx = fmaxf(a0, a1);
#pragma unroll
      for (int off = 8; off >= 1; off >>= 1)
        mx = fmaxf(mx, __shfl_xor(mx, off));
      const float mnew = fmaxf(rowmax[r], mx);
      const float corr = __expf(rowmax[r] - mnew);
      rowmax[r] = mnew;
      const float p0 = __expf(a0 - mnew), p1 = __expf(a1 - mnew);
      float rs = p0 + p1;
#pragma unroll
      for (int off = 8; off >= 1; off >>= 1) rs += __shfl_xor(rs, off);
      rowsum[r] = rowsum[r] * corr + rs;
#pragma unroll
      for (int j = 0; j < 4; ++j) o[j][r] *= corr;
      s0[r] = p0;
      s1[r] = p1;
    }

    // ---- C-layout -> A-layout for P via per-wave LDS scratch ----
    __bf16* pw = &sP[wid * (16 * 34)];
#pragma unroll
    for (int r = 0; r < 8; ++r) {
      pw[(hl * 8 + r) * 34 + nloc]      = (__bf16)s0[r];
      pw[(hl * 8 + r) * 34 + 16 + nloc] = (__bf16)s1[r];
    }
    v16bf pa = load_a_frag(pw, 34);

    // ---- O += P (16x32) * V (32x64) ----
#pragma unroll
    for (int j = 0; j < 4; ++j)
      o[j] = WMMA_BF16(pa, load_b_frag(&sV[cur][j * 16], LDH), o[j]);

    __syncthreads();
  }

  // ---- normalize and store (concat layout [b, s, h*64 + hd], bf16) ----
  float inv[8];
#pragma unroll
  for (int r = 0; r < 8; ++r) inv[r] = 1.0f / rowsum[r];
#pragma unroll
  for (int j = 0; j < 4; ++j)
#pragma unroll
    for (int r = 0; r < 8; ++r) {
      const int row = rowbase + qrow0 + hl * 8 + r;
      O[(size_t)row * DD + hoff + j * 16 + nloc] = (__bf16)(o[j][r] * inv[r]);
    }
}

// ---------------------------------------------------------------------------

extern "C" void kernel_launch(void* const* d_in, const int* in_sizes, int n_in,
                              void* d_out, int out_size, void* d_ws,
                              size_t ws_size, hipStream_t stream) {
  const float* q   = (const float*)d_in[0];
  const float* k   = (const float*)d_in[1];
  const float* v   = (const float*)d_in[2];
  const float* WQw = (const float*)d_in[3];
  const float* WQb = (const float*)d_in[4];
  const float* WKw = (const float*)d_in[5];
  const float* WKb = (const float*)d_in[6];
  const float* WVw = (const float*)d_in[7];
  const float* WVb = (const float*)d_in[8];
  const float* WOw = (const float*)d_in[9];
  const float* WOb = (const float*)d_in[10];
  float* out = (float*)d_out;

  const size_t MD = (size_t)MTOT * DD;  // 4M elems, bf16 -> 8 MB each
  __bf16* Qp = (__bf16*)d_ws;
  __bf16* Kp = Qp + MD;
  __bf16* Vp = Kp + MD;
  __bf16* AO = Vp + MD;   // total 32 MB of workspace

  dim3 blk(256);
  dim3 gproj(MTOT / TM, DD / TN);          // 32 x 8
  gemm_bias_kernel<float, __bf16><<<gproj, blk, 0, stream>>>(q, WQw, WQb, Qp,
                                                             MTOT, DD, DD);
  gemm_bias_kernel<float, __bf16><<<gproj, blk, 0, stream>>>(k, WKw, WKb, Kp,
                                                             MTOT, DD, DD);
  gemm_bias_kernel<float, __bf16><<<gproj, blk, 0, stream>>>(v, WVw, WVb, Vp,
                                                             MTOT, DD, DD);

  dim3 gattn(SS / QB, BB * HH);            // 16 x 32
  attention_kernel<<<gattn, blk, 0, stream>>>(Qp, Kp, Vp, AO);

  gemm_bias_kernel<__bf16, float><<<gproj, blk, 0, stream>>>(AO, WOw, WOb, out,
                                                             MTOT, DD, DD);
}
